// SelfAttention_59030030516896
// MI455X (gfx1250) — matrix operand order repaired and checked by reading, not compiled, not run on hardware
//
#include <hip/hip_runtime.h>
#include <hip/hip_fp16.h>

typedef __attribute__((ext_vector_type(16))) _Float16 v16h;
typedef __attribute__((ext_vector_type(8)))  _Float16 v8h;
typedef __attribute__((ext_vector_type(8)))  float    v8f;

#define NB 4
#define CC 256
#define CQ 32
#define HWP 4096   // 64*64 spatial positions

static __device__ __forceinline__ v8f wmma16(v16h a, v16h b, v8f c) {
  // D = A(16x32 f16) * B(32x16 f16) + C(16x16 f32)
  return __builtin_amdgcn_wmma_f32_16x16x32_f16(false, a, false, b, (short)0, c, false, false);
}

// swizzle of a position within a 32-element K-block so that lane-half h reads
// 16 contiguous halves: stored order [0-7][16-23][8-15][24-31]
static __device__ __forceinline__ int ksw(int pos) {
  return ((pos >> 3) & 1) * 16 + ((pos >> 4) & 1) * 8 + (pos & 7);
}

// LDS byte offset of a __shared__ pointer (flat LDS aperture: low 32 bits)
static __device__ __forceinline__ unsigned lds_off(const void* p) {
  return (unsigned)(uintptr_t)p;
}

// async 16B global -> LDS copy (per-lane), tracked by ASYNCcnt
static __device__ __forceinline__ void async_b128(unsigned lds, const void* g) {
  asm volatile("global_load_async_to_lds_b128 %0, %1, off"
               :: "v"(lds), "v"(g) : "memory");
}

// ---------------- kernel 1: convert weights fp32 -> fp16 ----------------
__global__ void k_cvt_w(const float* __restrict__ Wq, const float* __restrict__ Wk,
                        const float* __restrict__ Wv,
                        _Float16* __restrict__ whq, _Float16* __restrict__ whk,
                        _Float16* __restrict__ whv) {
  int i = blockIdx.x * blockDim.x + threadIdx.x;   // 65536 threads
  if (i < CQ * CC) { whq[i] = (_Float16)Wq[i]; whk[i] = (_Float16)Wk[i]; }
  whv[i] = (_Float16)Wv[i];                        // exactly 256*256
}

// ---------------- kernel 2: x[n][c][p] -> xh[n][p][c] f16, c contiguous ----
__global__ void k_xt(const float* __restrict__ x, _Float16* __restrict__ xh) {
  int t  = blockIdx.x * blockDim.x + threadIdx.x;  // 4*4096*32 threads
  int c8 = t & 31;            // which 8-channel chunk
  int p  = (t >> 5) & 4095;
  int n  = t >> 17;
  int c0 = c8 * 8;
  int pos = c0 & 31;
  int swstart = pos;
  const float* src = x + ((size_t)n * CC + c0) * HWP + p;   // lanes differ in c8 (stride 8*HWP), p fixed per wave
  v8h v;
#pragma unroll
  for (int k = 0; k < 8; ++k) v[k] = (_Float16)src[(size_t)k * HWP];
  _Float16* dst = xh + ((size_t)n * HWP + p) * CC + (c0 & ~31) + swstart;
  *(v8h*)dst = v;
}

// ---------------- kernel 3: Q/K projections (WMMA) ----------------------
// q[o][p] = sum_c Wq[o][c] x[c][p] + bq[o]; stored qT[n][p][sw(o)] f16
__global__ __launch_bounds__(128) void k_proj_qk(
    const _Float16* __restrict__ xh, const _Float16* __restrict__ whq,
    const _Float16* __restrict__ whk, const float* __restrict__ bq,
    const float* __restrict__ bk, _Float16* __restrict__ qT, _Float16* __restrict__ kT) {
  int gw   = (blockIdx.x * blockDim.x + threadIdx.x) >> 5;  // 4096 waves
  int lane = threadIdx.x & 31;
  int nl = lane & 15, h = lane >> 4;
  int b   = gw >> 10;
  int rem = gw & 1023;
  int pt  = rem >> 2;          // 256 position tiles
  int ot  = (rem >> 1) & 1;    // 2 output tiles (CQ=32)
  int sel = rem & 1;           // 0=q, 1=k
  const _Float16* W    = sel ? whk : whq;
  const float*    bias = sel ? bk  : bq;
  _Float16*       dst  = sel ? kT  : qT;
  int p0 = pt * 16;
  const _Float16* arow = xh + ((size_t)b * HWP + p0 + nl) * CC;  // A: M=p rows
  const _Float16* brow = W + (size_t)(ot * 16 + nl) * CC;        // B: N=o cols
  v8f acc = {};
#pragma unroll
  for (int kk = 0; kk < 8; ++kk) {
    v16h a  = *(const v16h*)(arow + kk * 32 + h * 16);
    v16h bm = *(const v16h*)(brow + kk * 32 + h * 16);
    acc = wmma16(a, bm, acc);
  }
  int o = ot * 16 + nl;
  float bvv = bias[o];
  int sw = ksw(o);
#pragma unroll
  for (int r = 0; r < 8; ++r)
    dst[((size_t)b * HWP + p0 + r + 8 * h) * CQ + sw] = (_Float16)(acc[r] + bvv);
}

// ---------------- kernel 4: V projection (WMMA) -------------------------
// v[c][p] stored vh[n][c][p] f16
__global__ __launch_bounds__(128) void k_proj_v(
    const _Float16* __restrict__ xh, const _Float16* __restrict__ whv,
    const float* __restrict__ bvp, _Float16* __restrict__ vh) {
  int gw   = (blockIdx.x * blockDim.x + threadIdx.x) >> 5;  // 16384 waves
  int lane = threadIdx.x & 31;
  int nl = lane & 15, h = lane >> 4;
  int b  = gw >> 12;
  int pt = (gw >> 4) & 255;
  int ct = gw & 15;
  int p0 = pt * 16;
  const _Float16* arow = xh + ((size_t)b * HWP + p0 + nl) * CC;
  const _Float16* brow = whv + (size_t)(ct * 16 + nl) * CC;
  v8f acc = {};
#pragma unroll
  for (int kk = 0; kk < 8; ++kk) {
    v16h a  = *(const v16h*)(arow + kk * 32 + h * 16);
    v16h bm = *(const v16h*)(brow + kk * 32 + h * 16);
    acc = wmma16(a, bm, acc);
  }
  int c = ct * 16 + nl;
  float bb = bvp[c];
#pragma unroll
  for (int r = 0; r < 8; ++r) {
    int p = p0 + r + 8 * h;
    vh[((size_t)b * CC + c) * HWP + p] = (_Float16)(acc[r] + bb);
  }
}

// ---------------- kernel 5: flash-style attention + residual ------------
// 256 threads = 8 waves; each wave owns one 16-row i-tile; all 8 waves share
// the same batch and j-schedule, so K/V j-tiles are staged in LDS once per
// block via double-buffered async global->LDS copies (ASYNCcnt).
__global__ __launch_bounds__(256) void k_attn(
    const _Float16* __restrict__ qT, const _Float16* __restrict__ kT,
    const _Float16* __restrict__ vh, const float* __restrict__ x,
    const float* __restrict__ gammap, float* __restrict__ out) {
  __shared__ __align__(128) _Float16 vbuf[2][CC * 32];   // 2 x 16KB
  __shared__ __align__(128) _Float16 kbuf[2][32 * 32];   // 2 x 2KB
  __shared__ __align__(64)  _Float16 pL[8][16 * 32];     // per-wave p tile [i][j]
  __shared__ float sS[8][16];                            // per-wave row sums
  int tid  = threadIdx.x;
  int wid  = tid >> 5;
  int lane = tid & 31;
  int nl = lane & 15, h = lane >> 4;
  int gw = blockIdx.x * 8 + wid;          // 1024 waves: 4 batches x 256 i-tiles
  int b  = gw >> 8;
  int i0 = (gw & 255) * 16;

  // A operand: q rows i0..i0+15 (whole CQ=32 fits one WMMA K)
  v16h qa = *(const v16h*)(qT + ((size_t)b * HWP + i0 + nl) * CQ + h * 16);
  const _Float16* kbase = kT + (size_t)b * HWP * CQ;
  const _Float16* vbase = vh + (size_t)b * CC * HWP;

  // ---- pass 1: row max of energy (K streams straight from L2) ----
  float vmax[8];
#pragma unroll
  for (int r = 0; r < 8; ++r) vmax[r] = -1e30f;
  for (int j = 0; j < HWP; j += 16) {
    __builtin_prefetch(kbase + (size_t)(j + 64 + nl) * CQ, 0, 1);
    v16h kb = *(const v16h*)(kbase + (size_t)(j + nl) * CQ + h * 16);
    v8f z = {};
    v8f e = wmma16(qa, kb, z);
#pragma unroll
    for (int r = 0; r < 8; ++r) vmax[r] = fmaxf(vmax[r], e[r]);
  }
#pragma unroll
  for (int r = 0; r < 8; ++r)
    for (int msk = 1; msk < 16; msk <<= 1)
      vmax[r] = fmaxf(vmax[r], __shfl_xor(vmax[r], msk, 32));
  // now every lane in half h holds row max for rows r+8h

  // cooperative async stage of one 32-j tile of K and V into LDS buffer `bi`
  auto issue_tile = [&](int bi, int j) {
    // V tile: 256 c-rows x 64B (plain j order) = 1024 x 16B chunks
#pragma unroll
    for (int q = 0; q < 4; ++q) {
      int chunk = tid + q * 256;
      int c = chunk >> 2, sub = chunk & 3;
      async_b128(lds_off(&vbuf[bi][c * 32 + sub * 8]),
                 vbase + (size_t)c * HWP + j + sub * 8);
    }
    // K tile: 32 rows x 64B = 128 x 16B chunks
    if (tid < 128) {
      int row = tid >> 2, sub = tid & 3;
      async_b128(lds_off(&kbuf[bi][row * 32 + sub * 8]),
                 kbase + (size_t)(j + row) * CQ + sub * 8);
    }
  };

  // ---- pass 2: P = exp(e - m), O += V * P^T, s += rowsum(P) ----
  v8f acc[16];
  v8f zero = {};
#pragma unroll
  for (int ct = 0; ct < 16; ++ct) acc[ct] = zero;
  float vs[8];
#pragma unroll
  for (int r = 0; r < 8; ++r) vs[r] = 0.0f;
  _Float16* pw = pL[wid];

  issue_tile(0, 0);
  for (int j = 0; j < HWP; j += 32) {
    int cur = (j >> 5) & 1;
    asm volatile("s_wait_asynccnt 0x0" ::: "memory");  // my chunks of buf[cur] landed
    __syncthreads();  // everyone's chunks landed; everyone done reading buf[cur^1]
    if (j + 32 < HWP) issue_tile(cur ^ 1, j + 32);

    v16h kb0 = *(const v16h*)&kbuf[cur][nl * 32 + h * 16];         // cols j..j+15
    v16h kb1 = *(const v16h*)&kbuf[cur][(16 + nl) * 32 + h * 16];  // cols j+16..j+31
    v8f e0 = wmma16(qa, kb0, zero);
    v8f e1 = wmma16(qa, kb1, zero);
#pragma unroll
    for (int r = 0; r < 8; ++r) {
      float p0 = __expf(e0[r] - vmax[r]);
      float p1 = __expf(e1[r] - vmax[r]);
      vs[r] += p0 + p1;
      pw[(r + 8 * h) * 32 + nl]      = (_Float16)p0;   // p[i][j]  (j sub 0)
      pw[(r + 8 * h) * 32 + 16 + nl] = (_Float16)p1;   // p[i][j]  (j sub 1)
    }
    // B operand: pT[K=j][N=i] == p[i][jblock], contiguous per lane
    v16h bp = *(const v16h*)(pw + nl * 32 + h * 16);
#pragma unroll
    for (int ct = 0; ct < 16; ++ct) {
      v16h av = *(const v16h*)&vbuf[cur][(ct * 16 + nl) * 32 + h * 16];
      acc[ct] = wmma16(av, bp, acc[ct]);
    }
  }

  // row sums -> all lanes need s[i = nl] via LDS exchange
#pragma unroll
  for (int r = 0; r < 8; ++r)
    for (int msk = 1; msk < 16; msk <<= 1)
      vs[r] += __shfl_xor(vs[r], msk, 32);
  if (nl == 0) {
#pragma unroll
    for (int r = 0; r < 8; ++r) sS[wid][r + 8 * h] = vs[r];
  }
  float sinv = 1.0f / sS[wid][nl];   // same-wave LDS RAW, in-order DS pipe
  float g = gammap[0];
#pragma unroll
  for (int ct = 0; ct < 16; ++ct) {
#pragma unroll
    for (int r = 0; r < 8; ++r) {
      size_t idx = ((size_t)b * CC + ct * 16 + r + 8 * h) * HWP + i0 + nl;
      out[idx] = g * acc[ct][r] * sinv + x[idx];
    }
  }
}

// ---------------- launcher ----------------------------------------------
extern "C" void kernel_launch(void* const* d_in, const int* in_sizes, int n_in,
                              void* d_out, int out_size, void* d_ws, size_t ws_size,
                              hipStream_t stream) {
  const float* x     = (const float*)d_in[0];
  const float* Wq    = (const float*)d_in[1];
  const float* bq    = (const float*)d_in[2];
  const float* Wk    = (const float*)d_in[3];
  const float* bk    = (const float*)d_in[4];
  const float* Wv    = (const float*)d_in[5];
  const float* bv    = (const float*)d_in[6];
  const float* gamma = (const float*)d_in[7];
  float* out = (float*)d_out;

  // workspace layout in f16 elements (~19 MB total)
  _Float16* ws  = (_Float16*)d_ws;
  _Float16* whq = ws;
  _Float16* whk = whq + (size_t)CQ * CC;
  _Float16* whv = whk + (size_t)CQ * CC;
  _Float16* xh  = whv + (size_t)CC * CC;
  _Float16* qT  = xh  + (size_t)NB * HWP * CC;
  _Float16* kT  = qT  + (size_t)NB * HWP * CQ;
  _Float16* vh  = kT  + (size_t)NB * HWP * CQ;

  k_cvt_w <<<256,  256, 0, stream>>>(Wq, Wk, Wv, whq, whk, whv);
  k_xt    <<<2048, 256, 0, stream>>>(x, xh);
  k_proj_qk<<<1024, 128, 0, stream>>>(xh, whq, whk, bq, bk, qT, kT);
  k_proj_v <<<4096, 128, 0, stream>>>(xh, whv, bv, vh);
  k_attn   <<<128,  256, 0, stream>>>(qT, kT, vh, x, gamma, out);
}
